// token_wise_choice_57475252355407
// MI455X (gfx1250) — matrix operand order repaired and checked by reading, not compiled, not run on hardware
//
#include <hip/hip_runtime.h>

// ---------------------------------------------------------------------------
// Problem constants (match reference)
// ---------------------------------------------------------------------------
constexpr int Bsz     = 4;
constexpr int T       = 2048;
constexpr int C       = 1024;
constexpr int HEADS   = 8;
constexpr int KEY_DIM = 512;
constexpr int KEY_NUM = 1024;
constexpr int KNN     = 32;
constexpr int HALF    = KEY_DIM / 2;          // 256
constexpr int QDIM    = HEADS * KEY_DIM;      // 4096
constexpr int BT      = Bsz * T;              // 8192
constexpr int HH      = 2 * HEADS;            // 16 head-halves

typedef __bf16 bf16;
typedef bf16  v16bf __attribute__((ext_vector_type(16)));
typedef bf16  v8bf  __attribute__((ext_vector_type(8)));
typedef float v8f   __attribute__((ext_vector_type(8)));

// ---------------------------------------------------------------------------
// CDNA5 async memory->LDS helpers (ASYNCcnt-tracked, no VGPR data round-trip)
// VDST VGPR carries the LDS byte offset; flat shared pointers have the LDS
// offset in their low 32 bits (flat LDS aperture), so truncation is exact.
// ---------------------------------------------------------------------------
__device__ __forceinline__ unsigned lds_off(const void* p) {
  return (unsigned)(uintptr_t)p;
}
__device__ __forceinline__ void async_copy16(const void* gsrc, unsigned ldst) {
  asm volatile("global_load_async_to_lds_b128 %0, %1, off"
               :: "v"(ldst), "v"(gsrc) : "memory");
}
__device__ __forceinline__ void wait_async0() {
  asm volatile("s_wait_asynccnt 0x0" ::: "memory");
}

// ---------------------------------------------------------------------------
// fp32 -> bf16 convert (weights / keys, done once per launch)
// ---------------------------------------------------------------------------
__global__ __launch_bounds__(256) void f32_to_bf16_kernel(
    const float* __restrict__ in, bf16* __restrict__ out, int n) {
  int i = blockIdx.x * 256 + threadIdx.x;
  if (i < n) out[i] = (bf16)in[i];
}

// ---------------------------------------------------------------------------
// Causal depthwise conv1d (K=3, left pad 2) + bias, output bf16 h[BT][C]
// out[t] = x[t-2]*w0 + x[t-1]*w1 + x[t]*w2 + b
// ---------------------------------------------------------------------------
__global__ __launch_bounds__(256) void conv_bf16_kernel(
    const float* __restrict__ x, const float* __restrict__ cw,
    const float* __restrict__ cb, bf16* __restrict__ h) {
  int idx = blockIdx.x * 256 + threadIdx.x;      // over BT*C
  if (idx >= BT * C) return;
  int c  = idx & (C - 1);
  int bt = idx >> 10;                            // C == 1024
  int t  = bt & (T - 1);                         // T == 2048
  float w0 = cw[c * 3 + 0], w1 = cw[c * 3 + 1], w2 = cw[c * 3 + 2];
  float v = x[idx] * w2;
  if (t >= 1) v += x[idx - C] * w1;
  if (t >= 2) v += x[idx - 2 * C] * w0;
  v += cb[c];
  h[idx] = (bf16)v;
}

// ---------------------------------------------------------------------------
// Helper: build 16x32 bf16 A-fragment from LDS (row-major, stride 32)
// Lane layout (contiguous K, matches B): lanes 0-15 row=lane, K = 0..15;
//                           lanes 16-31 row=lane-16, K = 16..31
// ---------------------------------------------------------------------------
__device__ __forceinline__ v16bf load_a_frag(const bf16* rowptr, int half) {
  const bf16* p = rowptr + half * 16;
  v8bf lo = *(const v8bf*)p;          // K base .. base+7
  v8bf hi = *(const v8bf*)(p + 8);    // K base+8 .. base+15
  return __builtin_shufflevector(lo, hi, 0, 1, 2, 3, 4, 5, 6, 7,
                                 8, 9, 10, 11, 12, 13, 14, 15);
}

// ---------------------------------------------------------------------------
// GEMM: q[BT][QDIM] = h[BT][C] @ lin_w[QDIM][C]^T + lin_b  (bf16 WMMA)
// Block tile 128x128, 8 waves (2 M x 4 N), wave tile 64x32 (4x2 accumulators)
// Double-buffered LDS tiles filled by global_load_async_to_lds_b128; WMMAs on
// buffer k overlap the async fill of buffer k+1 (s_wait_asynccnt + barrier).
// ---------------------------------------------------------------------------
__global__ __launch_bounds__(256) void qgemm_kernel(
    const bf16* __restrict__ hbf, const bf16* __restrict__ wbf,
    const float* __restrict__ lin_b, bf16* __restrict__ qbf) {
  __shared__ bf16 At[2][128 * 32];
  __shared__ bf16 Bt[2][128 * 32];
  const int tid  = threadIdx.x;
  const int lane = tid & 31;
  const int w    = tid >> 5;
  const int wm   = w >> 2;            // 0..1
  const int wn   = w & 3;             // 0..3
  const int half = lane >> 4;
  const int l16  = lane & 15;
  const int m0   = blockIdx.x * 128;
  const int n0   = blockIdx.y * 128;

  // per-thread tile segment (2x 16B per tile): rows/cols fixed across K-steps
  int seg0 = tid, seg1 = tid + 256;
  int r0 = seg0 >> 2, c0 = (seg0 & 3) << 3;
  int r1 = seg1 >> 2, c1 = (seg1 & 3) << 3;

  auto stage = [&](int buf, int k0) {
    async_copy16(&hbf[(size_t)(m0 + r0) * C + k0 + c0],
                 lds_off(&At[buf][r0 * 32 + c0]));
    async_copy16(&wbf[(size_t)(n0 + r0) * C + k0 + c0],
                 lds_off(&Bt[buf][r0 * 32 + c0]));
    async_copy16(&hbf[(size_t)(m0 + r1) * C + k0 + c1],
                 lds_off(&At[buf][r1 * 32 + c1]));
    async_copy16(&wbf[(size_t)(n0 + r1) * C + k0 + c1],
                 lds_off(&Bt[buf][r1 * 32 + c1]));
  };

  v8f acc[4][2] = {};

  stage(0, 0);                        // prologue fill of buffer 0
  wait_async0();
  __syncthreads();

  for (int k0 = 0; k0 < C; k0 += 32) {
    const int buf = (k0 >> 5) & 1;
    if (k0 + 32 < C) {
      stage(buf ^ 1, k0 + 32);        // async fill next buffer (overlaps WMMA)
      if (k0 + 64 < C) {              // L2 prefetch one tile further ahead
        __builtin_prefetch(&hbf[(size_t)(m0 + r0) * C + k0 + 64], 0, 1);
        __builtin_prefetch(&wbf[(size_t)(n0 + r0) * C + k0 + 64], 0, 1);
      }
    }

    v16bf afr[4], bfr[2];
    #pragma unroll
    for (int mf = 0; mf < 4; ++mf)
      afr[mf] = load_a_frag(&At[buf][(wm * 64 + mf * 16 + l16) * 32], half);
    #pragma unroll
    for (int nf = 0; nf < 2; ++nf)    // B frag: contiguous 16 K per lane
      bfr[nf] =
          *(const v16bf*)&Bt[buf][(wn * 32 + nf * 16 + l16) * 32 + half * 16];

    #pragma unroll
    for (int mf = 0; mf < 4; ++mf)
      #pragma unroll
      for (int nf = 0; nf < 2; ++nf)
        acc[mf][nf] = __builtin_amdgcn_wmma_f32_16x16x32_bf16(
            false, afr[mf], false, bfr[nf], (short)0, acc[mf][nf],
            false, false);

    wait_async0();                    // own async fills landed in LDS
    __syncthreads();                  // everyone done reading buf / filling
  }

  // epilogue: + lin_b, store bf16 q
  #pragma unroll
  for (int nf = 0; nf < 2; ++nf) {
    int col = n0 + wn * 32 + nf * 16 + l16;
    float bias = lin_b[col];
    #pragma unroll
    for (int mf = 0; mf < 4; ++mf) {
      #pragma unroll
      for (int v = 0; v < 8; ++v) {
        int row = m0 + wm * 64 + mf * 16 + v + half * 8;
        qbf[(size_t)row * QDIM + col] = (bf16)(acc[mf][nf][v] + bias);
      }
    }
  }
}

// ---------------------------------------------------------------------------
// Stage-1: scores + fused top-32.
// grid = (BT/16, 16 head-halves). Block: 16 rows x 1024 keys, K=256.
// Wave w owns key columns [w*128, w*128+128) -> 8 accumulators; B fragments
// load straight from global (keys are row-major [n][k] = B-frag shape).
// Scores land in 64KB LDS (A tile aliases its head), then per-row 32x
// max-extraction with jax tie-breaking (smaller index on equal value).
// ---------------------------------------------------------------------------
__global__ __launch_bounds__(256) void score_topk_kernel(
    const bf16* __restrict__ qbf, const bf16* __restrict__ kbf,
    float* __restrict__ sc_st, int* __restrict__ id_st) {
  __shared__ float smem[16 * 1024];             // 64KB score slab
  bf16* At = reinterpret_cast<bf16*>(smem);     // 1KB alias during GEMM phase

  const int tid  = threadIdx.x;
  const int lane = tid & 31;
  const int w    = tid >> 5;
  const int half = lane >> 4;
  const int l16  = lane & 15;
  const int m0   = blockIdx.x * 16;
  const int hh   = blockIdx.y;                  // head-half 0..15
  const int koff = hh * HALF;                   // (= l*512 + j*256)
  const size_t keybase = (size_t)hh * KEY_NUM * HALF;

  v8f acc[8] = {};

  for (int k0 = 0; k0 < HALF; k0 += 32) {
    if (tid < 64) {                             // 16 rows x 32 K of q -> LDS
      int row = tid >> 2;
      int c8  = (tid & 3) << 3;
      async_copy16(&qbf[(size_t)(m0 + row) * QDIM + koff + k0 + c8],
                   lds_off(&At[row * 32 + c8]));
    }
    wait_async0();
    __syncthreads();
    v16bf a = load_a_frag(&At[l16 * 32], half);
    #pragma unroll
    for (int nf = 0; nf < 8; ++nf) {
      int n = w * 128 + nf * 16 + l16;
      v16bf b = *(const v16bf*)&kbf[keybase + (size_t)n * HALF + k0 + half * 16];
      acc[nf] = __builtin_amdgcn_wmma_f32_16x16x32_bf16(
          false, a, false, b, (short)0, acc[nf], false, false);
    }
    __syncthreads();
  }

  // accumulators -> LDS score slab
  #pragma unroll
  for (int nf = 0; nf < 8; ++nf) {
    int col = w * 128 + nf * 16 + l16;
    #pragma unroll
    for (int v = 0; v < 8; ++v)
      smem[(v + half * 8) * 1024 + col] = acc[nf][v];
  }
  __syncthreads();

  // top-32 per row; wave w owns rows 2w and 2w+1
  for (int rr = 0; rr < 2; ++rr) {
    const int r = w * 2 + rr;
    float* srow = &smem[r * 1024];
    for (int it = 0; it < KNN; ++it) {
      float best  = -__builtin_inff();
      int   bestc = 0x7fffffff;
      #pragma unroll
      for (int j = 0; j < 32; ++j) {
        int c = lane + (j << 5);
        float v = srow[c];
        if (v > best || (v == best && c < bestc)) { best = v; bestc = c; }
      }
      #pragma unroll
      for (int off = 16; off > 0; off >>= 1) {
        float ov = __shfl_xor(best, off, 32);
        int   oc = __shfl_xor(bestc, off, 32);
        if (ov > best || (ov == best && oc < bestc)) { best = ov; bestc = oc; }
      }
      if (lane == (bestc & 31)) srow[bestc] = -__builtin_inff();
      if (lane == 0) {
        size_t o = ((size_t)(m0 + r) * HH + hh) * KNN + it;
        sc_st[o] = best;
        id_st[o] = bestc;
      }
    }
  }
}

// ---------------------------------------------------------------------------
// Stage-2: cartesian 32x32 combine + top-32, one wave per (token, head).
// Lane i holds sc1[i]+sc2[j], j=0..31 in registers; 32 extractions give
// sorted-descending output; idx = id1*KEY_NUM + id2.
// ---------------------------------------------------------------------------
__global__ __launch_bounds__(256) void combine_topk_kernel(
    const float* __restrict__ sc_st, const int* __restrict__ id_st,
    float* __restrict__ score_out, int* __restrict__ idx_out) {
  const int tid  = threadIdx.x;
  const int lane = tid & 31;
  const int w    = tid >> 5;
  const int gw   = blockIdx.x * 8 + w;          // 0 .. BT*HEADS-1
  const int m    = gw >> 3;
  const int l    = gw & 7;

  const size_t b1 = ((size_t)m * HH + l * 2 + 0) * KNN;
  const size_t b2 = ((size_t)m * HH + l * 2 + 1) * KNN;
  float s1 = sc_st[b1 + lane];
  float s2 = sc_st[b2 + lane];
  int   i1 = id_st[b1 + lane];
  int   i2 = id_st[b2 + lane];

  float cmb[32];
  #pragma unroll
  for (int j = 0; j < 32; ++j) cmb[j] = s1 + __shfl(s2, j, 32);

  unsigned used = 0u;
  float keep_s = 0.0f;
  int   keep_i = 0;

  for (int it = 0; it < KNN; ++it) {
    float best  = -__builtin_inff();
    int   bestc = 0x7fffffff;
    #pragma unroll
    for (int j = 0; j < 32; ++j) {              // per-lane c ascending: > keeps
      if (used & (1u << j)) continue;           // smallest index on ties
      float v = cmb[j];
      if (v > best) { best = v; bestc = (lane << 5) | j; }
    }
    #pragma unroll
    for (int off = 16; off > 0; off >>= 1) {
      float ov = __shfl_xor(best, off, 32);
      int   oc = __shfl_xor(bestc, off, 32);
      if (ov > best || (ov == best && oc < bestc)) { best = ov; bestc = oc; }
    }
    if (lane == (bestc >> 5)) used |= 1u << (bestc & 31);
    int ii = __shfl(i1, bestc >> 5, 32);
    int jj = __shfl(i2, bestc & 31, 32);
    if (lane == it) { keep_s = best; keep_i = ii * KEY_NUM + jj; }
  }

  size_t o = (size_t)m * (HEADS * KNN) + l * KNN + lane;
  score_out[o] = keep_s;
  idx_out[o]   = keep_i;
}

// ---------------------------------------------------------------------------
// Host launcher
// ---------------------------------------------------------------------------
extern "C" void kernel_launch(void* const* d_in, const int* in_sizes, int n_in,
                              void* d_out, int out_size, void* d_ws,
                              size_t ws_size, hipStream_t stream) {
  const float* x      = (const float*)d_in[0];
  const float* conv_w = (const float*)d_in[1];
  const float* conv_b = (const float*)d_in[2];
  const float* lin_w  = (const float*)d_in[3];
  const float* lin_b  = (const float*)d_in[4];
  const float* keys   = (const float*)d_in[5];

  char* ws = (char*)d_ws;
  bf16* hbf   = (bf16*)ws;  ws += (size_t)BT * C * sizeof(bf16);
  bf16* wbf   = (bf16*)ws;  ws += (size_t)QDIM * C * sizeof(bf16);
  bf16* kbf   = (bf16*)ws;  ws += (size_t)HH * KEY_NUM * HALF * sizeof(bf16);
  bf16* qbf   = (bf16*)ws;  ws += (size_t)BT * QDIM * sizeof(bf16);
  float* sc_st = (float*)ws; ws += (size_t)BT * HH * KNN * sizeof(float);
  int*   id_st = (int*)ws;   ws += (size_t)BT * HH * KNN * sizeof(int);

  const int n_w = QDIM * C;             // 4,194,304
  const int n_k = HH * KEY_NUM * HALF;  // 4,194,304
  f32_to_bf16_kernel<<<(n_w + 255) / 256, 256, 0, stream>>>(lin_w, wbf, n_w);
  f32_to_bf16_kernel<<<(n_k + 255) / 256, 256, 0, stream>>>(keys, kbf, n_k);

  conv_bf16_kernel<<<(BT * C) / 256, 256, 0, stream>>>(x, conv_w, conv_b, hbf);

  qgemm_kernel<<<dim3(BT / 128, QDIM / 128), 256, 0, stream>>>(
      hbf, wbf, lin_b, qbf);

  score_topk_kernel<<<dim3(BT / 16, HH), 256, 0, stream>>>(
      qbf, kbf, sc_st, id_st);

  float* score_out = (float*)d_out;
  int*   idx_out   = (int*)(score_out + (size_t)BT * HEADS * KNN);
  combine_topk_kernel<<<(BT * HEADS) / 8, 256, 0, stream>>>(
      sc_st, id_st, score_out, idx_out);
}
